// Aligned_Feature_Aggregation_80985903333675
// MI455X (gfx1250) — matrix operand order repaired and checked by reading, not compiled, not run on hardware
//
#include <hip/hip_runtime.h>
#include <hip/hip_bf16.h>
#include <stdint.h>

typedef __attribute__((ext_vector_type(16))) _Float16 v16h;
typedef __attribute__((ext_vector_type(8)))  float    v8f;

#define BB   2
#define NP   8192
#define CF   64
#define OC   256
#define KNN  16
#define FCIN 4144   /* 259*16 */
#define BIGF 3.0e38f

// ---------------------------------------------------------------------------
// Kernel 1: transpose xyz2 / feature_2 to point-major, 3-NN upsample of
// feature_1 from xyz_1 onto xyz_2 -> in_cat[p][0:64]=up1, [64:128]=feat2.
// Also emits an f16 copy for the WMMA A-matrix path.
// ---------------------------------------------------------------------------
__global__ void k_pre_upsample(const float* __restrict__ xyz1,
                               const float* __restrict__ xyz2,
                               const float* __restrict__ feat1,
                               const float* __restrict__ feat2,
                               float* __restrict__ in_cat,      // [B][N][128] f32
                               _Float16* __restrict__ in_cat_h, // [B][N][128] f16
                               float* __restrict__ xyz2t)       // [B][N][3]
{
    int p = blockIdx.x * blockDim.x + threadIdx.x;
    if (p >= BB * NP) return;
    int b = p / NP, n = p % NP;

    const float* xb2 = xyz2 + (size_t)b * 3 * NP;
    float qx = xb2[n], qy = xb2[NP + n], qz = xb2[2 * NP + n];
    xyz2t[3 * p] = qx; xyz2t[3 * p + 1] = qy; xyz2t[3 * p + 2] = qz;

    const float* f2 = feat2 + (size_t)b * CF * NP;
    float*    dst  = in_cat   + (size_t)p * 128;
    _Float16* dsth = in_cat_h + (size_t)p * 128;
    for (int c = 0; c < CF; ++c) {
        float v = f2[(size_t)c * NP + n];
        dst[64 + c] = v; dsth[64 + c] = (_Float16)v;
    }

    // brute-force 3 nearest neighbors in xyz1
    const float* xb1 = xyz1 + (size_t)b * 3 * NP;
    float d0 = BIGF, d1 = BIGF, d2 = BIGF;
    int   i0 = 0,    i1 = 0,    i2 = 0;
    for (int s = 0; s < NP; ++s) {
        float dx = xb1[s] - qx, dy = xb1[NP + s] - qy, dz = xb1[2 * NP + s] - qz;
        float d = dx * dx + dy * dy + dz * dz;
        if (d < d2) {
            if (d < d0)      { d2 = d1; i2 = i1; d1 = d0; i1 = i0; d0 = d; i0 = s; }
            else if (d < d1) { d2 = d1; i2 = i1; d1 = d;  i1 = s; }
            else             { d2 = d;  i2 = s; }
        }
    }
    float w0 = 1.0f / fmaxf(sqrtf(d0), 1e-10f);
    float w1 = 1.0f / fmaxf(sqrtf(d1), 1e-10f);
    float w2 = 1.0f / fmaxf(sqrtf(d2), 1e-10f);
    float wsum = w0 + w1 + w2; w0 /= wsum; w1 /= wsum; w2 /= wsum;

    const float* f1 = feat1 + (size_t)b * CF * NP;
    for (int c = 0; c < CF; ++c) {
        const float* row = f1 + (size_t)c * NP;
        float v = w0 * row[i0] + w1 * row[i1] + w2 * row[i2];
        dst[c] = v; dsth[c] = (_Float16)v;
    }
}

// ---------------------------------------------------------------------------
// Kernel 2: self 16-NN per point (register insertion sort).
// ---------------------------------------------------------------------------
__global__ void k_knn16(const float* __restrict__ xyz2t, int* __restrict__ knn_idx)
{
    int p = blockIdx.x * blockDim.x + threadIdx.x;
    if (p >= BB * NP) return;
    int b = p / NP;
    const float* base = xyz2t + (size_t)b * NP * 3;
    float qx = xyz2t[3 * p], qy = xyz2t[3 * p + 1], qz = xyz2t[3 * p + 2];

    float bd[KNN]; int bi[KNN];
#pragma unroll
    for (int j = 0; j < KNN; ++j) { bd[j] = BIGF; bi[j] = 0; }

    for (int m = 0; m < NP; ++m) {
        float dx = base[3 * m] - qx, dy = base[3 * m + 1] - qy, dz = base[3 * m + 2] - qz;
        float d = dx * dx + dy * dy + dz * dz;
        if (d < bd[KNN - 1]) {
            bd[KNN - 1] = d; bi[KNN - 1] = m;
#pragma unroll
            for (int j = KNN - 1; j > 0; --j) {
                if (bd[j] < bd[j - 1]) {
                    float td = bd[j]; bd[j] = bd[j - 1]; bd[j - 1] = td;
                    int   ti = bi[j]; bi[j] = bi[j - 1]; bi[j - 1] = ti;
                }
            }
        }
    }
    int* outp = knn_idx + (size_t)p * KNN;
#pragma unroll
    for (int j = 0; j < KNN; ++j) outp[j] = bi[j];
}

// ---------------------------------------------------------------------------
// Kernel 2b: prepack conv1 weights into WMMA B-fragment order (f16).
// wfrag[((ct*4+kk)*32+lane)*16 + 2v..] = W^T pairs per ISA 32x16 B layout.
// ---------------------------------------------------------------------------
__global__ __launch_bounds__(128) void k_pack_w(const float* __restrict__ w,
                                                _Float16* __restrict__ wfrag)
{
    int ct   = blockIdx.x;           // 16 channel tiles
    int kk   = threadIdx.x >> 5;     // 4 K-steps (one wave each)
    int lane = threadIdx.x & 31;
    int r    = lane & 15;
    int half = lane >> 4;
    _Float16* dst = wfrag + ((size_t)((ct * 4 + kk) * 32 + lane)) * 16;
#pragma unroll
    for (int v = 0; v < 8; ++v) {
        int k0 = 2 * (v & 3) + 16 * (v >> 2) + 8 * half + 32 * kk;
        float2 f = *(const float2*)(w + (size_t)(ct * 16 + r) * 128 + k0);
        dst[2 * v]     = (_Float16)f.x;
        dst[2 * v + 1] = (_Float16)f.y;
    }
}

// ---------------------------------------------------------------------------
// Kernel 3: conv1 (256x128) + BN + LeakyReLU as WMMA GEMM.
// Each wave: 16 points x 2 channel-tiles, K=128 in 4 steps -> 8 wmma,
// A-fragment reused for both accumulators.
// ---------------------------------------------------------------------------
__global__ __launch_bounds__(128) void k_conv1_wmma(
    const _Float16* __restrict__ in_cat_h, const _Float16* __restrict__ wfrag,
    const float* __restrict__ bias,  const float* __restrict__ gamma,
    const float* __restrict__ beta,  const float* __restrict__ mean,
    const float* __restrict__ var,   _Float16* __restrict__ nf_h)
{
    int tile = blockIdx.x;                   // point tile (N/16)
    int b    = blockIdx.z;
    int wid  = threadIdx.x >> 5;
    int ct0  = blockIdx.y * 8 + wid * 2;     // first of 2 channel tiles
    int lane = threadIdx.x & 31;
    int r    = lane & 15;
    int half = lane >> 4;

    const _Float16* arow = in_cat_h + ((size_t)(b * NP) + tile * 16 + r) * 128;

    v8f acc0 = {}, acc1 = {};
#pragma unroll
    for (int kk = 0; kk < 4; ++kk) {
        v16h a;
        uint32_t* aw = (uint32_t*)&a;
#pragma unroll
        for (int v = 0; v < 8; ++v) {
            int kb = 2 * (v & 3) + 16 * (v >> 2) + 8 * half + 32 * kk;
            aw[v] = *(const uint32_t*)(arow + kb);
        }
        v16h b0f, b1f;
        uint32_t* b0w = (uint32_t*)&b0f;
        uint32_t* b1w = (uint32_t*)&b1f;
        const uint4* w0p = (const uint4*)(wfrag + ((size_t)((ct0 * 4 + kk) * 32 + lane)) * 16);
        const uint4* w1p = (const uint4*)(wfrag + ((size_t)(((ct0 + 1) * 4 + kk) * 32 + lane)) * 16);
        uint4 q0 = w0p[0], q1 = w0p[1];
        uint4 q2 = w1p[0], q3 = w1p[1];
        b0w[0] = q0.x; b0w[1] = q0.y; b0w[2] = q0.z; b0w[3] = q0.w;
        b0w[4] = q1.x; b0w[5] = q1.y; b0w[6] = q1.z; b0w[7] = q1.w;
        b1w[0] = q2.x; b1w[1] = q2.y; b1w[2] = q2.z; b1w[3] = q2.w;
        b1w[4] = q3.x; b1w[5] = q3.y; b1w[6] = q3.z; b1w[7] = q3.w;
        acc0 = __builtin_amdgcn_wmma_f32_16x16x32_f16(false, a, false, b0f,
                                                      (short)0, acc0, false, false);
        acc1 = __builtin_amdgcn_wmma_f32_16x16x32_f16(false, a, false, b1f,
                                                      (short)0, acc1, false, false);
    }

#pragma unroll
    for (int t2 = 0; t2 < 2; ++t2) {
        int ch = (ct0 + t2) * 16 + r;
        float cb = bias[ch];
        float sc = gamma[ch] * rsqrtf(var[ch] + 1e-5f);
        float mu = mean[ch], be = beta[ch];
#pragma unroll
        for (int v = 0; v < 8; ++v) {
            int pm = tile * 16 + v + 8 * half;    // D row = point
            float val = (t2 == 0 ? acc0[v] : acc1[v]) + cb;
            val = (val - mu) * sc + be;
            val = (val >= 0.f) ? val : 0.1f * val;
            nf_h[((size_t)(b * NP) + pm) * 256 + ch] = (_Float16)val;
        }
    }
}

// ---------------------------------------------------------------------------
// Kernel 4: per-point WeightNet + agg einsum (WMMA, 17 c-tiles) + fc(6x4144).
// Neighbor-feature gather uses CDNA5 async global->LDS b128 DMA (ASYNCcnt).
// Ph layout: cols [0,256) = features, [256,259) = rel xyz, [259,272) = pad.
// ---------------------------------------------------------------------------
__global__ __launch_bounds__(256) void k_agg_fc(
    const float* __restrict__ xyz2t, const int* __restrict__ knn_idx,
    const _Float16* __restrict__ nf_h,
    const float* __restrict__ w0, const float* __restrict__ b0,
    const float* __restrict__ w1, const float* __restrict__ b1,
    const float* __restrict__ w2, const float* __restrict__ b2,
    const float* __restrict__ fc_w, const float* __restrict__ fc_b,
    float* __restrict__ warped1, float* __restrict__ warped2)
{
    __shared__ _Float16 Ph[32][272];   // row stride 544B (16B multiple)
    __shared__ _Float16 Wh[32][16];
    __shared__ float    rel[16][3];
    __shared__ int      nidx[16];
    __shared__ float    red[256][6];
    __shared__ float    outacc[6];

    int tid = threadIdx.x;
    int p   = blockIdx.x;
    int b   = p / NP;

    uint32_t* phz = (uint32_t*)&Ph[0][0];
    for (int t = tid; t < (32 * 272) / 2; t += 256) phz[t] = 0u;
    uint32_t* whz = (uint32_t*)&Wh[0][0];
    if (tid < (32 * 16) / 2) whz[tid] = 0u;

    if (tid < 16) {
        int nb = knn_idx[(size_t)p * KNN + tid];
        nidx[tid] = nb;
        int gq = b * NP + nb;
        rel[tid][0] = xyz2t[3 * gq]     - xyz2t[3 * p];
        rel[tid][1] = xyz2t[3 * gq + 1] - xyz2t[3 * p + 1];
        rel[tid][2] = xyz2t[3 * gq + 2] - xyz2t[3 * p + 2];
    }
    __syncthreads();  // zero-fill + nidx/rel visible before async DMA lands

    // async gather of 16 neighbor feature rows (512B each) into LDS:
    // 512 x b128 transfers, 2 per thread, tracked by ASYNCcnt.
    for (int t = tid; t < 16 * 32; t += 256) {
        int k = t >> 5, q = t & 31;
        const _Float16* gsrc = nf_h + ((size_t)(b * NP) + nidx[k]) * 256 + q * 8;
        unsigned loff = (unsigned)(uintptr_t)(__attribute__((address_space(3))) _Float16*)&Ph[k][q * 8];
        uint64_t gp = (uint64_t)(uintptr_t)gsrc;
        asm volatile("global_load_async_to_lds_b128 %0, %1, off"
                     :: "v"(loff), "v"(gp) : "memory");
    }

    if (tid < 48) { int k = tid / 3, c = tid % 3; Ph[k][256 + c] = (_Float16)rel[k][c]; }

    if (tid < 16) {  // WeightNet MLP 3->8->8->16 for neighbor k = tid
        float gx = rel[tid][0], gy = rel[tid][1], gz = rel[tid][2];
        float h0[8], h1[8];
#pragma unroll
        for (int i = 0; i < 8; ++i)
            h0[i] = fmaxf(w0[i * 3] * gx + w0[i * 3 + 1] * gy + w0[i * 3 + 2] * gz + b0[i], 0.f);
#pragma unroll
        for (int i = 0; i < 8; ++i) {
            float s = b1[i];
#pragma unroll
            for (int j = 0; j < 8; ++j) s += w1[i * 8 + j] * h0[j];
            h1[i] = fmaxf(s, 0.f);
        }
#pragma unroll
        for (int i = 0; i < 16; ++i) {
            float s = b2[i];
#pragma unroll
            for (int j = 0; j < 8; ++j) s += w2[i * 8 + j] * h1[j];
            Wh[tid][i] = (_Float16)fmaxf(s, 0.f);
        }
    }

#if __has_builtin(__builtin_amdgcn_s_wait_asynccnt)
    __builtin_amdgcn_s_wait_asynccnt(0);
#else
    asm volatile("s_wait_asynccnt 0x0" ::: "memory");
#endif
    __syncthreads();

    int wave = tid >> 5;
    int lane = tid & 31;
    int r    = lane & 15;
    int half = lane >> 4;

    float acc6[6] = {0.f, 0.f, 0.f, 0.f, 0.f, 0.f};
    for (int t = wave; t < 17; t += 8) {        // wave-uniform loop (EXEC full)
        int cg = t * 16 + r;                    // A row
        v16h a, bb;
#pragma unroll
        for (int v = 0; v < 8; ++v) {
            int kb = 2 * (v & 3) + 16 * (v >> 2) + 8 * half;
            a[2 * v]     = Ph[kb][cg];
            a[2 * v + 1] = Ph[kb + 1][cg];
            int k0 = 2 * (v & 3) + 16 * (v >> 2) + 8 * half;
            bb[2 * v]     = Wh[k0][r];
            bb[2 * v + 1] = Wh[k0 + 1][r];
        }
        v8f c = {};
        c = __builtin_amdgcn_wmma_f32_16x16x32_f16(false, a, false, bb,
                                                   (short)0, c, false, false);
#pragma unroll
        for (int v = 0; v < 8; ++v) {
            int crow = t * 16 + v + 8 * half;
            if (crow < 259) {
                // remap to reference order: c = [xyz(3), feat(256)]
                int refc = (crow < 256) ? (crow + 3) : (crow - 256);
                int fidx = refc * 16 + r;       // flatten order c*16 + w
#pragma unroll
                for (int o = 0; o < 6; ++o)
                    acc6[o] += fc_w[(size_t)o * FCIN + fidx] * c[v];
            }
        }
    }

#pragma unroll
    for (int o = 0; o < 6; ++o) red[tid][o] = acc6[o];
    __syncthreads();
    if (tid < 6) {                 // fixed-order deterministic reduction
        float s = 0.f;
        for (int i = 0; i < 256; ++i) s += red[i][tid];
        outacc[tid] = s;
    }
    __syncthreads();
    if (tid < 3) {
        float base = xyz2t[3 * p + tid];
        warped1[3 * p + tid] = outacc[tid]     + fc_b[tid]     + base;
        warped2[3 * p + tid] = outacc[3 + tid] + fc_b[3 + tid] + base;
    }
}

// ---------------------------------------------------------------------------
// Kernel 5: 1-NN vs warped_1 / warped_2, gather up1/feat2, add -> output.
// ---------------------------------------------------------------------------
__global__ void k_final(const float* __restrict__ xyz2t,
                        const float* __restrict__ warped1,
                        const float* __restrict__ warped2,
                        const float* __restrict__ in_cat,
                        float* __restrict__ out)
{
    int p = blockIdx.x * blockDim.x + threadIdx.x;
    if (p >= BB * NP) return;
    int b = p / NP;
    float qx = xyz2t[3 * p], qy = xyz2t[3 * p + 1], qz = xyz2t[3 * p + 2];
    const float* w1b = warped1 + (size_t)b * NP * 3;
    const float* w2b = warped2 + (size_t)b * NP * 3;

    float bd1 = BIGF, bd2 = BIGF; int i1 = 0, i2 = 0;
    for (int m = 0; m < NP; ++m) {
        float dx = w1b[3 * m] - qx, dy = w1b[3 * m + 1] - qy, dz = w1b[3 * m + 2] - qz;
        float d = dx * dx + dy * dy + dz * dz;
        if (d < bd1) { bd1 = d; i1 = m; }
        dx = w2b[3 * m] - qx; dy = w2b[3 * m + 1] - qy; dz = w2b[3 * m + 2] - qz;
        d = dx * dx + dy * dy + dz * dz;
        if (d < bd2) { bd2 = d; i2 = m; }
    }
    const float* up = in_cat + ((size_t)(b * NP) + i1) * 128;       // up1 part
    const float* f2 = in_cat + ((size_t)(b * NP) + i2) * 128 + 64;  // feat2 part
    float* op = out + (size_t)p * CF;
    for (int c = 0; c < CF; ++c) op[c] = up[c] + f2[c];
}

// ---------------------------------------------------------------------------
extern "C" void kernel_launch(void* const* d_in, const int* in_sizes, int n_in,
                              void* d_out, int out_size, void* d_ws, size_t ws_size,
                              hipStream_t stream)
{
    const float* xyz_1     = (const float*)d_in[0];
    const float* xyz_2     = (const float*)d_in[1];
    const float* feature_1 = (const float*)d_in[2];
    const float* feature_2 = (const float*)d_in[3];
    const float* conv1_w   = (const float*)d_in[4];
    const float* conv1_b   = (const float*)d_in[5];
    const float* bn_gamma  = (const float*)d_in[6];
    const float* bn_beta   = (const float*)d_in[7];
    const float* bn_mean   = (const float*)d_in[8];
    const float* bn_var    = (const float*)d_in[9];
    const float* wn_w0     = (const float*)d_in[10];
    const float* wn_b0     = (const float*)d_in[11];
    const float* wn_w1     = (const float*)d_in[12];
    const float* wn_b1     = (const float*)d_in[13];
    const float* wn_w2     = (const float*)d_in[14];
    const float* wn_b2     = (const float*)d_in[15];
    const float* fc_w      = (const float*)d_in[16];
    const float* fc_b      = (const float*)d_in[17];
    (void)in_sizes; (void)n_in; (void)out_size; (void)ws_size;

    char* ws = (char*)d_ws;
    size_t off = 0;
    auto carve = [&](size_t bytes) -> void* {
        void* ptr = ws + off;
        off = (off + bytes + 255) & ~(size_t)255;
        return ptr;
    };
    float*    in_cat   = (float*)carve((size_t)BB * NP * 128 * sizeof(float));
    _Float16* in_cat_h = (_Float16*)carve((size_t)BB * NP * 128 * sizeof(_Float16));
    float*    xyz2t    = (float*)carve((size_t)BB * NP * 3 * sizeof(float));
    int*      knn_idx  = (int*)carve((size_t)BB * NP * KNN * sizeof(int));
    _Float16* nf_h     = (_Float16*)carve((size_t)BB * NP * 256 * sizeof(_Float16));
    _Float16* wfrag    = (_Float16*)carve((size_t)16 * 4 * 32 * 16 * sizeof(_Float16));
    float*    warped1  = (float*)carve((size_t)BB * NP * 3 * sizeof(float));
    float*    warped2  = (float*)carve((size_t)BB * NP * 3 * sizeof(float));

    int total = BB * NP;
    k_pre_upsample<<<dim3((total + 255) / 256), dim3(256), 0, stream>>>(
        xyz_1, xyz_2, feature_1, feature_2, in_cat, in_cat_h, xyz2t);
    k_pack_w<<<dim3(16), dim3(128), 0, stream>>>(conv1_w, wfrag);
    k_knn16<<<dim3((total + 255) / 256), dim3(256), 0, stream>>>(xyz2t, knn_idx);
    k_conv1_wmma<<<dim3(NP / 16, OC / 128, BB), dim3(128), 0, stream>>>(
        in_cat_h, wfrag, conv1_b, bn_gamma, bn_beta, bn_mean, bn_var, nf_h);
    k_agg_fc<<<dim3(total), dim3(256), 0, stream>>>(
        xyz2t, knn_idx, nf_h, wn_w0, wn_b0, wn_w1, wn_b1, wn_w2, wn_b2,
        fc_w, fc_b, warped1, warped2);
    k_final<<<dim3((total + 255) / 256), dim3(256), 0, stream>>>(
        xyz2t, warped1, warped2, in_cat, (float*)d_out);
}
